// GNNDynamics_34050500722853
// MI455X (gfx1250) — matrix operand order repaired and checked by reading, not compiled, not run on hardware
//
#include <hip/hip_runtime.h>

typedef __attribute__((ext_vector_type(16))) __bf16 v16bf;
typedef __attribute__((ext_vector_type(8)))  __bf16 v8bf;
typedef __attribute__((ext_vector_type(4)))  __bf16 v4bf;
typedef __attribute__((ext_vector_type(8)))  float  v8f;

#define LATENT   512
#define B_TOTAL  65536
#define E_ACT    2016
#define M_TILE   32
#define M_TILES  2          // 16-row wmma tiles per block
#define THREADS  256
#define NWAVE    8

// bf16 workspace element offsets (weights transposed to [N][Kpad])
#define W1T_OFF   0          // [256][672]  (K=656 padded to 672)
#define W1T_K     672
#define W2T_OFF   172032     // [512][256]
#define W2T_K     256
#define RW1T_OFF  303104     // [256][512]
#define RW1T_K    512
#define RW2T_OFF  434176     // [640][256] (N=601 padded to 640)
#define RW2T_K    256
#define WS_ELEMS  598016

// LDS strides (bf16 elements)
#define H_STRIDE  672
#define H1_STRIDE 264
#define Z_STRIDE  520

// ---------------------------------------------------------------------------
// Kernel 0: f32 -> bf16 weight conversion + transpose (+ zero padding)
// ---------------------------------------------------------------------------
__global__ __launch_bounds__(256) void convert_weights(
    const float* __restrict__ w1, const float* __restrict__ w2,
    const float* __restrict__ rw1, const float* __restrict__ rw2,
    __bf16* __restrict__ ws)
{
    int idx = blockIdx.x * 256 + threadIdx.x;
    if (idx >= WS_ELEMS) return;
    float v = 0.0f;
    if (idx < W2T_OFF) {                       // w1t [256][672] from w1[656][256]
        int n = idx / W1T_K, k = idx % W1T_K;
        if (k < 656) v = w1[k * 256 + n];
    } else if (idx < RW1T_OFF) {               // w2t [512][256] from w2[256][512]
        int t = idx - W2T_OFF;
        int n = t / W2T_K, k = t % W2T_K;
        v = w2[k * 512 + n];
    } else if (idx < RW2T_OFF) {               // rw1t [256][512] from rw1[512][256]
        int t = idx - RW1T_OFF;
        int n = t / RW1T_K, k = t % RW1T_K;
        v = rw1[k * 256 + n];
    } else {                                   // rw2t [640][256] from rw2[256][601]
        int t = idx - RW2T_OFF;
        int n = t / RW2T_K, k = t % RW2T_K;
        if (n < 601) v = rw2[k * 601 + n];
    }
    ws[idx] = (__bf16)v;
}

// ---------------------------------------------------------------------------
// Fragment helpers (layouts per CDNA5 ISA 7.12.2, wave32)
// ---------------------------------------------------------------------------
// A (16x32 bf16): lane = {half,row}; VGPR 0-3 -> K = kb+8*half+{0..7},
//                 VGPR 4-7 -> K = kb+16+8*half+{0..7}  (two ds_load_b128)
__device__ __forceinline__ v16bf load_a_frag(const __bf16* s, int stride, int m,
                                             int kb, int lcol, int half)
{
    const __bf16* p = s + (m * 16 + lcol) * stride + kb + half * 8;
    v16bf a;
    ((v8bf*)&a)[0] = *(const v8bf*)p;
    ((v8bf*)&a)[1] = *(const v8bf*)(p + 16);
    return a;
}

// B (32x16 bf16) from N-major (transposed) weights: col = lane&15,
// elems 0-7 -> K = kb+8*half+{0..7}, 8-15 -> K = kb+16+8*half+{0..7} (2x b128).
__device__ __forceinline__ v16bf load_b_frag(const __bf16* wt, int Kpad, int ncol,
                                             int kb, int half)
{
    const __bf16* p = wt + (size_t)ncol * Kpad + kb + half * 8;
    v16bf b;
    ((v8bf*)&b)[0] = *(const v8bf*)p;
    ((v8bf*)&b)[1] = *(const v8bf*)(p + 16);
    return b;
}

// ---------------------------------------------------------------------------
// One fused GEMM stage over the block's 32-row tile.
// Each wave owns pairs of 16-col output tiles; acc = 2 m-tiles x 2 n-tiles.
// ---------------------------------------------------------------------------
template <bool RELU, bool TO_LDS, bool TO_GLOBAL>
__device__ __forceinline__ void gemm_stage(
    const __bf16* __restrict__ sIn, int strideIn,
    const __bf16* __restrict__ wt, int Kpad,
    const float* __restrict__ bias,
    int nPairs, int kTiles,
    __bf16* __restrict__ sOut, int strideOut,
    float* __restrict__ gOut, int gStride, int Nvalid,
    int wave, int lcol, int half, int row0)
{
    for (int p = wave; p < nPairs; p += NWAVE) {
        const int n0 = p * 32;
        v8f zero = {0.f, 0.f, 0.f, 0.f, 0.f, 0.f, 0.f, 0.f};
        v8f acc[M_TILES][2];
        #pragma unroll
        for (int m = 0; m < M_TILES; ++m) { acc[m][0] = zero; acc[m][1] = zero; }

        for (int kc = 0; kc < kTiles; ++kc) {
            const int kb = kc * 32;
            v16bf b0 = load_b_frag(wt, Kpad, n0 + lcol,      kb, half);
            v16bf b1 = load_b_frag(wt, Kpad, n0 + 16 + lcol, kb, half);
            #pragma unroll
            for (int m = 0; m < M_TILES; ++m) {
                v16bf a = load_a_frag(sIn, strideIn, m, kb, lcol, half);
                acc[m][0] = __builtin_amdgcn_wmma_f32_16x16x32_bf16(
                    false, a, false, b0, (short)0, acc[m][0], false, false);
                acc[m][1] = __builtin_amdgcn_wmma_f32_16x16x32_bf16(
                    false, a, false, b1, (short)0, acc[m][1], false, false);
            }
        }

        // Epilogue: bias + relu; bf16 -> LDS for next stage, f32 -> HBM.
        #pragma unroll
        for (int t = 0; t < 2; ++t) {
            const int col = n0 + t * 16 + lcol;
            const bool valid = col < Nvalid;
            float bv = bias[valid ? col : 0];   // branchless, uniform-ish load
            if (!valid) bv = 0.f;
            #pragma unroll
            for (int m = 0; m < M_TILES; ++m) {
                #pragma unroll
                for (int r = 0; r < 8; ++r) {
                    const int row = m * 16 + r + half * 8;
                    float v = acc[m][t][r] + bv;
                    if (RELU) v = fmaxf(v, 0.f);
                    if (TO_LDS) sOut[row * strideOut + col] = (__bf16)v;
                    if (TO_GLOBAL && valid)
                        gOut[(size_t)(row0 + row) * gStride + col] = v;
                }
            }
        }
    }
}

// ---------------------------------------------------------------------------
// Fused dynamics kernel: 32 rows/block, full 4-GEMM chain through LDS.
// ---------------------------------------------------------------------------
__global__ __launch_bounds__(256) void fused_dyn(
    const float* __restrict__ z, const int* __restrict__ action_idx,
    const float* __restrict__ node_emb, const float* __restrict__ color_emb,
    const float* __restrict__ b1, const float* __restrict__ b2,
    const float* __restrict__ rb1, const float* __restrict__ rb2,
    const __bf16* __restrict__ ws,
    float* __restrict__ outZ, float* __restrict__ outR)
{
    // Static LDS: provably non-null addrspace(3) objects -> clean ds_* codegen.
    __shared__ __align__(16) __bf16 shA[M_TILE * H_STRIDE];   // h / z_next
    __shared__ __align__(16) __bf16 shB[M_TILE * H1_STRIDE];  // h1 / h2
    __shared__ int shI[M_TILE], shJ[M_TILE], shC[M_TILE];

    const int tid  = threadIdx.x;
    const int lane = tid & 31;
    const int wave = tid >> 5;
    const int lcol = lane & 15;
    const int half = lane >> 4;
    const int row0 = blockIdx.x * M_TILE;

    // --- triangular action decode (closed form + fixup), one thread per row ---
    if (tid < M_TILE) {
        int a = action_idx[row0 + tid];
        a = a < 0 ? 0 : (a > E_ACT ? E_ACT : a);
        int I = 0, J = 0, c = 1;
        if (a < E_ACT) {
            int i = (int)(63.5f - sqrtf(63.5f * 63.5f - 2.0f * (float)a));
            i = i < 0 ? 0 : (i > 62 ? 62 : i);
            while (i < 62 && (64 * (i + 1) - ((i + 1) * (i + 2)) / 2) <= a) ++i;
            while (i > 0  && (64 * i - (i * (i + 1)) / 2) > a) --i;
            I = i;
            J = a - (64 * i - (i * (i + 1)) / 2) + i + 1;
            c = 0;
        }
        shI[tid] = I; shJ[tid] = J; shC[tid] = c;
    }

    // --- z tile -> shA cols [0,512), f32 -> bf16, float4-vectorized ---
    const float4* z4 = (const float4*)(z + (size_t)row0 * LATENT);
    for (int idx = tid; idx < M_TILE * 128; idx += THREADS) {
        float4 v = z4[idx];
        int r = idx >> 7, c4 = (idx & 127) << 2;
        v4bf pk = {(__bf16)v.x, (__bf16)v.y, (__bf16)v.z, (__bf16)v.w};
        *(v4bf*)(shA + r * H_STRIDE + c4) = pk;
    }
    __syncthreads();

    // --- embedding gather -> shA cols [512,672) (zero-padded tail) ---
    for (int idx = tid; idx < M_TILE * 160; idx += THREADS) {
        int r = idx / 160, c = idx % 160;
        float v = 0.f;
        if (c < 64)       v = node_emb[shI[r] * 64 + c];
        else if (c < 128) v = node_emb[shJ[r] * 64 + (c - 64)];
        else if (c < 144) v = color_emb[shC[r] * 16 + (c - 128)];
        shA[r * H_STRIDE + 512 + c] = (__bf16)v;
    }
    __syncthreads();

    // Stage 1: h1 = relu(h @ W1 + b1)            [32,672]x[672,256]
    gemm_stage<true, true, false>(shA, H_STRIDE, ws + W1T_OFF, W1T_K, b1,
                                  256 / 32, 672 / 32, shB, H1_STRIDE,
                                  nullptr, 0, 256, wave, lcol, half, row0);
    __syncthreads();

    // Stage 2: z_next = h1 @ W2 + b2 (-> HBM f32 and LDS bf16)
    gemm_stage<false, true, true>(shB, H1_STRIDE, ws + W2T_OFF, W2T_K, b2,
                                  512 / 32, 256 / 32, shA, Z_STRIDE,
                                  outZ, 512, 512, wave, lcol, half, row0);
    __syncthreads();

    // Stage 3: h2 = relu(z_next @ RW1 + rb1)
    gemm_stage<true, true, false>(shA, Z_STRIDE, ws + RW1T_OFF, RW1T_K, rb1,
                                  256 / 32, 512 / 32, shB, H1_STRIDE,
                                  nullptr, 0, 256, wave, lcol, half, row0);
    __syncthreads();

    // Stage 4: r_logits = h2 @ RW2 + rb2 (N padded 601 -> 640, masked store)
    gemm_stage<false, false, true>(shB, H1_STRIDE, ws + RW2T_OFF, RW2T_K, rb2,
                                   640 / 32, 256 / 32, nullptr, 0,
                                   outR, 601, 601, wave, lcol, half, row0);
}

// ---------------------------------------------------------------------------
extern "C" void kernel_launch(void* const* d_in, const int* in_sizes, int n_in,
                              void* d_out, int out_size, void* d_ws, size_t ws_size,
                              hipStream_t stream)
{
    const float* z         = (const float*)d_in[0];
    const int*   act       = (const int*)  d_in[1];
    const float* node_emb  = (const float*)d_in[2];
    const float* color_emb = (const float*)d_in[3];
    const float* w1        = (const float*)d_in[4];
    const float* b1        = (const float*)d_in[5];
    const float* w2        = (const float*)d_in[6];
    const float* b2        = (const float*)d_in[7];
    const float* rw1       = (const float*)d_in[8];
    const float* rb1       = (const float*)d_in[9];
    const float* rw2       = (const float*)d_in[10];
    const float* rb2       = (const float*)d_in[11];

    __bf16* ws   = (__bf16*)d_ws;
    float*  outZ = (float*)d_out;
    float*  outR = outZ + (size_t)B_TOTAL * LATENT;

    convert_weights<<<(WS_ELEMS + 255) / 256, 256, 0, stream>>>(w1, w2, rw1, rw2, ws);

    fused_dyn<<<B_TOTAL / M_TILE, THREADS, 0, stream>>>(
        z, act, node_emb, color_emb, b1, b2, rb1, rb2, ws, outZ, outR);
}
